// StitchEncoder_19069654794601
// MI455X (gfx1250) — matrix operand order repaired and checked by reading, not compiled, not run on hardware
//
#include <hip/hip_runtime.h>

typedef __attribute__((ext_vector_type(16))) __bf16 v16bf;
typedef __attribute__((ext_vector_type(8)))  __bf16 v8bf;
typedef __attribute__((ext_vector_type(2)))  __bf16 v2bf;
typedef __attribute__((ext_vector_type(8)))  float  v8f;
typedef __attribute__((ext_vector_type(4)))  int    v4i;

#define B_   256
#define T_   100
#define N_   768
#define N2_  1536
#define P_   768

#define BM   128   // covers all T=100 rows in one tile (padded to 128)
#define BN   64
#define BK   32
#define LDSK 40    // padded LDS row stride in elements (80B = 5*16B, keeps v8bf aligned)

#if defined(__gfx1250__) && __has_builtin(__builtin_amdgcn_global_load_async_to_lds_b128) && \
    __has_builtin(__builtin_amdgcn_s_wait_asynccnt)
#define USE_ASYNC 1
#else
#define USE_ASYNC 0
#endif

#if USE_ASYNC
// builtin signature (from clang diagnostic): arg0 = v4i in addrspace(1)
#define AS_GLOBAL_V4I(p) ((__attribute__((address_space(1))) v4i*)(uintptr_t)(p))
#define AS_LOCAL_V4I(p)  ((__attribute__((address_space(3))) v4i*)(unsigned)(uintptr_t)(p))
#endif

#define SHUF16(lo, hi) __builtin_shufflevector(lo, hi, 0,1,2,3,4,5,6,7,8,9,10,11,12,13,14,15)

// A fragment (16x32 bf16): lanes 0-15 hold row r, K{0..7,16..23}; lanes 16-31 K{8..15,24..31}
__device__ __forceinline__ v16bf load_a_frag(const __bf16* A, int row, int half) {
    const __bf16* ap = A + row * LDSK;
    v8bf lo = *(const v8bf*)(ap + (half ? 8  : 0));
    v8bf hi = *(const v8bf*)(ap + (half ? 24 : 16));
    return SHUF16(lo, hi);
}
// B fragment (32x16 bf16, column-major LDS): lane holds column, K{0..7,16..23} / K{8..15,24..31} by half
__device__ __forceinline__ v16bf load_b_frag(const __bf16* Bq, int col, int half) {
    const __bf16* bp = Bq + col * LDSK + half * 8;
    v8bf lo = *(const v8bf*)bp;
    v8bf hi = *(const v8bf*)(bp + 16);
    return SHUF16(lo, hi);
}

#define WMMA_BF16(a, bmat, c) \
    __builtin_amdgcn_wmma_f32_16x16x32_bf16(false, a, false, bmat, (short)0, c, false, false)

// ---------------------------------------------------------------------------
// Kernel 1: h[b,t,m] = softsign( x[b,t,:] @ W_stitch[eid[b]] + b_stitch[eid[b]] )
// ---------------------------------------------------------------------------
__global__ __launch_bounds__(256) void stitch_kernel(
    const float* __restrict__ x,      // [B,T,N] f32
    const float* __restrict__ Ws,     // [E,N,2N] f32
    const float* __restrict__ bsv,    // [E,2N] f32
    const int*   __restrict__ eid,    // [B]
    __bf16*      __restrict__ h)      // [B,T,2N] bf16 (scratch)
{
    __shared__ __align__(16) __bf16 As[2][BM * LDSK];   // 2 x 10240 B
    __shared__ __align__(16) __bf16 Bs[2][BN * LDSK];   // 2 x  5120 B

    const int tid  = threadIdx.x;
    const int col0 = blockIdx.x * BN;
    const int b    = blockIdx.y;
    const int e    = eid[b];

    const float* xb = x  + (size_t)b * T_ * N_;
    const float* W  = Ws + (size_t)e * N_ * N2_;

    const int lane = tid & 31;
    const int w    = tid >> 5;
    const int wm   = w & 3;        // 4 row strips of 32
    const int wn   = w >> 2;       // 2 col strips of 32
    const int half = lane >> 4;
    const int r16  = lane & 15;

    // B transpose mapping: thread owns K-row pair (kr2, kr2+1) x 4 cols
    const int kr2 = (tid >> 4) << 1;        // 0,2,...,30
    const int n4  = (tid & 15) << 2;        // 0,4,...,60

    float4 ra[4];   // A staging: 128x32 f32 = 1024 float4 / 256 thr = 4 each
    float4 rb[2];   // B staging: two K-rows of one float4 column chunk

    auto load_tile = [&](int k0) {
        #pragma unroll
        for (int j = 0; j < 4; ++j) {
            int i   = tid + j * 256;
            int row = i >> 3;
            int c4  = (i & 7) << 2;
            int gm  = row < T_ ? row : (T_ - 1);   // clamp: rows >= T discarded in epilogue
            ra[j] = *(const float4*)(xb + (size_t)gm * N_ + k0 + c4);
        }
        const float* p = W + (size_t)(k0 + kr2) * N2_ + col0 + n4;
        rb[0] = *(const float4*)p;
        rb[1] = *(const float4*)(p + N2_);
    };
    auto store_tile = [&](int buf) {
        __bf16* A  = As[buf];
        __bf16* Bq = Bs[buf];
        #pragma unroll
        for (int j = 0; j < 4; ++j) {
            int i   = tid + j * 256;
            int row = i >> 3;
            int c4  = (i & 7) << 2;
            __bf16* p = A + row * LDSK + c4;
            p[0] = (__bf16)ra[j].x; p[1] = (__bf16)ra[j].y;
            p[2] = (__bf16)ra[j].z; p[3] = (__bf16)ra[j].w;
        }
        // packed K-pair transpose stores: one b32 per column
        v2bf p0 = { (__bf16)rb[0].x, (__bf16)rb[1].x };
        v2bf p1 = { (__bf16)rb[0].y, (__bf16)rb[1].y };
        v2bf p2 = { (__bf16)rb[0].z, (__bf16)rb[1].z };
        v2bf p3 = { (__bf16)rb[0].w, (__bf16)rb[1].w };
        *(v2bf*)(Bq + (n4 + 0) * LDSK + kr2) = p0;
        *(v2bf*)(Bq + (n4 + 1) * LDSK + kr2) = p1;
        *(v2bf*)(Bq + (n4 + 2) * LDSK + kr2) = p2;
        *(v2bf*)(Bq + (n4 + 3) * LDSK + kr2) = p3;
    };

    v8f acc[2][2] = {};

    load_tile(0);
    store_tile(0);
    __syncthreads();

    const int NK = N_ / BK;   // 24
    for (int kk = 0; kk < NK; ++kk) {
        const int cur = kk & 1;
        if (kk + 1 < NK) load_tile((kk + 1) * BK);

        const __bf16* A  = As[cur];
        const __bf16* Bq = Bs[cur];
        v16bf a0 = load_a_frag(A, wm * 32 + r16,      half);
        v16bf a1 = load_a_frag(A, wm * 32 + 16 + r16, half);
        v16bf b0 = load_b_frag(Bq, wn * 32 + r16,      half);
        v16bf b1 = load_b_frag(Bq, wn * 32 + 16 + r16, half);
        acc[0][0] = WMMA_BF16(a0, b0, acc[0][0]);
        acc[0][1] = WMMA_BF16(a0, b1, acc[0][1]);
        acc[1][0] = WMMA_BF16(a1, b0, acc[1][0]);
        acc[1][1] = WMMA_BF16(a1, b1, acc[1][1]);

        if (kk + 1 < NK) store_tile(1 - cur);
        __syncthreads();
    }

    #pragma unroll
    for (int mf = 0; mf < 2; ++mf) {
        #pragma unroll
        for (int nf = 0; nf < 2; ++nf) {
            int colg = col0 + wn * 32 + nf * 16 + r16;
            float bias = bsv[(size_t)e * N2_ + colg];
            #pragma unroll
            for (int i = 0; i < 8; ++i) {
                int row = wm * 32 + mf * 16 + half * 8 + i;   // C VGPR i -> M = i / i+8
                if (row < T_) {
                    float v = acc[mf][nf][i] + bias;
                    v = v / (1.0f + fabsf(v));
                    h[((size_t)b * T_ + row) * N2_ + colg] = (__bf16)v;
                }
            }
        }
    }
}

// ---------------------------------------------------------------------------
// Kernel 2: out[b,t,p] = h[b,t,:] @ W_proj[eid[b]] + b_proj[eid[b]]
// A-tile is a straight bf16 copy -> use async global->LDS when available.
// ---------------------------------------------------------------------------
__global__ __launch_bounds__(256) void proj_kernel(
    const __bf16* __restrict__ h,     // [B,T,2N] bf16 (scratch)
    const float*  __restrict__ Wp,    // [E,2N,P] f32
    const float*  __restrict__ bpv,   // [E,P] f32
    const int*    __restrict__ eid,   // [B]
    float*        __restrict__ out)   // [B,T,P] f32
{
    __shared__ __align__(16) __bf16 As[2][BM * LDSK];
    __shared__ __align__(16) __bf16 Bs[2][BN * LDSK];

    const int tid  = threadIdx.x;
    const int col0 = blockIdx.x * BN;
    const int b    = blockIdx.y;
    const int e    = eid[b];

    const __bf16* hb = h  + (size_t)b * T_ * N2_;
    const float*  W  = Wp + (size_t)e * N2_ * P_;

    const int lane = tid & 31;
    const int w    = tid >> 5;
    const int wm   = w & 3;
    const int wn   = w >> 2;
    const int half = lane >> 4;
    const int r16  = lane & 15;

    const int kr2 = (tid >> 4) << 1;
    const int n4  = (tid & 15) << 2;

    float4 rb[2];
#if !USE_ASYNC
    v8bf ra[2];
#endif

    // --- A tile: 128x32 bf16 = 512 x 16B chunks / 256 thr = 2 each ---
    auto issue_a = [&](int k0, int buf) {
        #pragma unroll
        for (int j = 0; j < 2; ++j) {
            int i   = tid + j * 256;
            int row = i >> 2;
            int c8  = (i & 3) << 3;
            int gm  = row < T_ ? row : (T_ - 1);
            const __bf16* gp = hb + (size_t)gm * N2_ + k0 + c8;
            __bf16*       lp = As[buf] + row * LDSK + c8;
#if USE_ASYNC
            // global -> LDS, no VGPR round trip; tracked by ASYNCcnt
            __builtin_amdgcn_global_load_async_to_lds_b128(
                AS_GLOBAL_V4I(gp), AS_LOCAL_V4I(lp), 0, 0);
#else
            ra[j] = *(const v8bf*)gp;
            (void)lp;
#endif
        }
    };
#if !USE_ASYNC
    auto store_a = [&](int buf) {
        #pragma unroll
        for (int j = 0; j < 2; ++j) {
            int i   = tid + j * 256;
            int row = i >> 2;
            int c8  = (i & 3) << 3;
            *(v8bf*)(As[buf] + row * LDSK + c8) = ra[j];
        }
    };
#endif
    auto load_b = [&](int k0) {
        const float* p = W + (size_t)(k0 + kr2) * P_ + col0 + n4;
        rb[0] = *(const float4*)p;
        rb[1] = *(const float4*)(p + P_);
    };
    auto store_b = [&](int buf) {
        __bf16* Bq = Bs[buf];
        v2bf p0 = { (__bf16)rb[0].x, (__bf16)rb[1].x };
        v2bf p1 = { (__bf16)rb[0].y, (__bf16)rb[1].y };
        v2bf p2 = { (__bf16)rb[0].z, (__bf16)rb[1].z };
        v2bf p3 = { (__bf16)rb[0].w, (__bf16)rb[1].w };
        *(v2bf*)(Bq + (n4 + 0) * LDSK + kr2) = p0;
        *(v2bf*)(Bq + (n4 + 1) * LDSK + kr2) = p1;
        *(v2bf*)(Bq + (n4 + 2) * LDSK + kr2) = p2;
        *(v2bf*)(Bq + (n4 + 3) * LDSK + kr2) = p3;
    };

    v8f acc[2][2] = {};

    // prologue: tile 0
    issue_a(0, 0);
    load_b(0);
    store_b(0);
#if USE_ASYNC
    __builtin_amdgcn_s_wait_asynccnt(0);
#else
    store_a(0);
#endif
    __syncthreads();

    const int NK = N2_ / BK;   // 48
    for (int kk = 0; kk < NK; ++kk) {
        const int cur = kk & 1;
        const bool has_next = (kk + 1 < NK);
        // buf[1-cur] reads all retired at previous barrier -> safe to refill now
        if (has_next) {
            issue_a((kk + 1) * BK, 1 - cur);
            load_b((kk + 1) * BK);
        }

        const __bf16* A  = As[cur];
        const __bf16* Bq = Bs[cur];
        v16bf a0 = load_a_frag(A, wm * 32 + r16,      half);
        v16bf a1 = load_a_frag(A, wm * 32 + 16 + r16, half);
        v16bf b0 = load_b_frag(Bq, wn * 32 + r16,      half);
        v16bf b1 = load_b_frag(Bq, wn * 32 + 16 + r16, half);
        acc[0][0] = WMMA_BF16(a0, b0, acc[0][0]);
        acc[0][1] = WMMA_BF16(a0, b1, acc[0][1]);
        acc[1][0] = WMMA_BF16(a1, b0, acc[1][0]);
        acc[1][1] = WMMA_BF16(a1, b1, acc[1][1]);

        if (has_next) {
            store_b(1 - cur);
#if !USE_ASYNC
            store_a(1 - cur);
#endif
        }
#if USE_ASYNC
        __builtin_amdgcn_s_wait_asynccnt(0);
#endif
        __syncthreads();
    }

    #pragma unroll
    for (int mf = 0; mf < 2; ++mf) {
        #pragma unroll
        for (int nf = 0; nf < 2; ++nf) {
            int colg = col0 + wn * 32 + nf * 16 + r16;
            float bias = bpv[(size_t)e * P_ + colg];
            #pragma unroll
            for (int i = 0; i < 8; ++i) {
                int row = wm * 32 + mf * 16 + half * 8 + i;
                if (row < T_) {
                    out[((size_t)b * T_ + row) * P_ + colg] = acc[mf][nf][i] + bias;
                }
            }
        }
    }
}

// ---------------------------------------------------------------------------
// Launcher. Inputs (dict order): x, W_stitch, b_stitch, W_proj, b_proj, eid.
// d_ws holds the bf16 intermediate h: 256*100*1536*2 = 78.6 MB.
// ---------------------------------------------------------------------------
extern "C" void kernel_launch(void* const* d_in, const int* in_sizes, int n_in,
                              void* d_out, int out_size, void* d_ws, size_t ws_size,
                              hipStream_t stream) {
    const float* x   = (const float*)d_in[0];
    const float* Ws  = (const float*)d_in[1];
    const float* bs  = (const float*)d_in[2];
    const float* Wp  = (const float*)d_in[3];
    const float* bp  = (const float*)d_in[4];
    const int*   eid = (const int*)d_in[5];

    __bf16* h   = (__bf16*)d_ws;
    float*  out = (float*)d_out;

    dim3 block(256);
    dim3 g1(N2_ / BN, B_);   // 24 x 256 blocks
    dim3 g2(P_  / BN, B_);   // 12 x 256 blocks

    stitch_kernel<<<g1, block, 0, stream>>>(x, Ws, bs, eid, h);
    proj_kernel  <<<g2, block, 0, stream>>>(h, Wp, bp, eid, out);
}
